// GCNConvModel_57294863729415
// MI455X (gfx1250) — matrix operand order repaired and checked by reading, not compiled, not run on hardware
//
#include <hip/hip_runtime.h>

#define N_NODES 100000
#define N_EDGES 1600000
#define N_GRAPHS 256
#define EPSV 1e-5f

typedef __attribute__((ext_vector_type(16))) _Float16 v16h;
typedef __attribute__((ext_vector_type(8)))  float    v8f;

// ---------------------------------------------------------------- utilities
__global__ void k_zero(float* __restrict__ p, int n) {
    int i = blockIdx.x * blockDim.x + threadIdx.x;
    int st = gridDim.x * blockDim.x;
    for (; i < n; i += st) p[i] = 0.0f;
}

// ------------------------------------------------------------- edge MLP: ew
__global__ void k_edge_mlp(const float* __restrict__ ea,
                           const float* __restrict__ w1, const float* __restrict__ b1,
                           const float* __restrict__ w2, const float* __restrict__ b2,
                           float* __restrict__ ew) {
    __shared__ float sw1[128], sb1[16], sw2[16];
    int tid = threadIdx.x;
    if (tid < 128) sw1[tid] = w1[tid];
    if (tid < 16) { sb1[tid] = b1[tid]; sw2[tid] = w2[tid]; }
    __syncthreads();
    int e = blockIdx.x * 256 + tid;
    if (e >= N_EDGES) return;
    const float4* p = (const float4*)(ea + (size_t)e * 8);
    float4 v0 = p[0], v1 = p[1];
    float a[8] = {v0.x, v0.y, v0.z, v0.w, v1.x, v1.y, v1.z, v1.w};
    float o = b2[0];
#pragma unroll
    for (int j = 0; j < 16; ++j) {
        float s = sb1[j];
#pragma unroll
        for (int k = 0; k < 8; ++k) s += sw1[j * 8 + k] * a[k];
        s = fmaxf(s, 0.0f);
        o += sw2[j] * s;
    }
    ew[e] = o;
}

// ------------------------------------------------- degree / counts / dinv
__global__ void k_deg(const int* __restrict__ tgt, const float* __restrict__ ew,
                      float* __restrict__ deg) {
    int e = blockIdx.x * 256 + threadIdx.x;
    if (e < N_EDGES) atomicAdd(&deg[tgt[e]], ew[e]);
}
__global__ void k_cnt(const int* __restrict__ batch, float* __restrict__ cnt) {
    int n = blockIdx.x * 256 + threadIdx.x;
    if (n < N_NODES) atomicAdd(&cnt[batch[n]], 1.0f);
}
__global__ void k_cntfix(float* __restrict__ cnt) {
    int g = threadIdx.x;
    if (g < N_GRAPHS) cnt[g] = fmaxf(cnt[g], 1.0f);
}
__global__ void k_dinv(const float* __restrict__ deg, float* __restrict__ dinv) {
    int n = blockIdx.x * 256 + threadIdx.x;
    if (n < N_NODES) {
        float d = deg[n] + 1.0f;
        dinv[n] = (d > 0.0f) ? rsqrtf(d) : 0.0f;
    }
}

// ------------------------------------------------------- WMMA GEMM  C = A*W
// A: [N,K] f32 row-major, W: [K,128] f32 row-major, C: [N,128] f32.
// Block = 256 threads (8 waves), 64 rows x 128 cols per block.
// wave -> rowt = wave&3 (16-row strip), colh = wave>>2 (64-col half, 4 tiles).
template <int K>
__global__ __launch_bounds__(256) void gemm_wmma(const float* __restrict__ A,
                                                 const float* __restrict__ W,
                                                 float* __restrict__ C, int N) {
    __shared__ _Float16 sA[64 * K];    // A tile, row-major [64][K]
    __shared__ _Float16 sWt[128 * K];  // W transposed [col][K]
    const int tid  = threadIdx.x;
    const int lane = tid & 31;
    const int wave = tid >> 5;
    const int row0 = blockIdx.x * 64;

    // Stage W (coalesced read W[k][c], transposed LDS write)
    for (int idx = tid; idx < K * 128; idx += 256) {
        int k = idx >> 7, c = idx & 127;
        sWt[c * K + k] = (_Float16)W[idx];
    }
    // Stage A tile (zero-pad rows past N)
    for (int idx = tid; idx < 64 * K; idx += 256) {
        int r = idx / K, k = idx % K;
        int n = row0 + r;
        sA[idx] = (n < N) ? (_Float16)A[(size_t)n * K + k] : (_Float16)0.0f;
    }
    __syncthreads();

    const int rowt = wave & 3;
    const int colh = wave >> 2;
    const int m  = lane & 15;
    const int hi = lane >> 4;  // 0: lanes 0-15, 1: lanes 16-31

    v8f zero = {0.f, 0.f, 0.f, 0.f, 0.f, 0.f, 0.f, 0.f};
    v8f acc[4];
#pragma unroll
    for (int t = 0; t < 4; ++t) acc[t] = zero;

    const _Float16* arow = &sA[(rowt * 16 + m) * K];

#pragma unroll
    for (int kc = 0; kc < K / 32; ++kc) {
        const int kb = kc * 32;
        // A fragment: 16x32 f16. lanes 0-15: K kb+0..7, kb+16..23 ; lanes 16-31: +8
        v16h a;
        const _Float16* ap = arow + kb + hi * 8;
#pragma unroll
        for (int i = 0; i < 8; ++i) { a[i] = ap[i]; a[8 + i] = ap[16 + i]; }
#pragma unroll
        for (int t = 0; t < 4; ++t) {
            // B fragment: 32x16 f16. lane = column; lanes 0-15: K kb+0..7, kb+16..23 ; lanes 16-31: +8
            const int col = colh * 64 + t * 16 + m;
            const _Float16* bp = &sWt[col * K + kb + hi * 8];
            v16h b;
#pragma unroll
            for (int i = 0; i < 8; ++i) { b[i] = bp[i]; b[8 + i] = bp[16 + i]; }
            acc[t] = __builtin_amdgcn_wmma_f32_16x16x32_f16(
                false, a, false, b, (short)0, acc[t], false, false);
        }
    }

    // Store: C VGPR v -> row rbase + v (lanes 16-31: +8), col = lane%16 in tile.
    // Fast path for full blocks (all but the last grid block): no bounds checks.
    const int rbase = row0 + rowt * 16 + hi * 8;
    if (row0 + 64 <= N) {
#pragma unroll
        for (int t = 0; t < 4; ++t) {
            const int col = colh * 64 + t * 16 + m;
            float* cp = C + (size_t)rbase * 128 + col;
#pragma unroll
            for (int v = 0; v < 8; ++v) cp[(size_t)v * 128] = acc[t][v];
        }
    } else {
#pragma unroll
        for (int t = 0; t < 4; ++t) {
            const int col = colh * 64 + t * 16 + m;
#pragma unroll
            for (int v = 0; v < 8; ++v) {
                int r = rbase + v;
                if (r < N) C[(size_t)r * 128 + col] = acc[t][v];
            }
        }
    }
}

// -------------------------------------------- GCN aggregation (self + edges)
__global__ void k_agg_init(const float* __restrict__ hw, const float* __restrict__ dinv,
                           const float* __restrict__ bias, float* __restrict__ agg) {
    int i = blockIdx.x * 256 + threadIdx.x;
    if (i < N_NODES * 128) {
        int n = i >> 7, c = i & 127;
        float di = dinv[n];
        agg[i] = di * di * hw[i] + bias[c];
    }
}

__global__ void k_scatter(const int* __restrict__ src, const int* __restrict__ tgt,
                          const float* __restrict__ ew, const float* __restrict__ dinv,
                          const float* __restrict__ hw, float* __restrict__ agg) {
    int e = blockIdx.x * 8 + (threadIdx.x >> 5);
    int lane = threadIdx.x & 31;
    if (e >= N_EDGES) return;
    int s = src[e], t = tgt[e];
    float coef = dinv[s] * ew[e] * dinv[t];
    const float4 v = *(const float4*)(hw + (size_t)s * 128 + lane * 4);
    float* ap = agg + (size_t)t * 128 + lane * 4;
    atomicAdd(ap + 0, coef * v.x);
    atomicAdd(ap + 1, coef * v.y);
    atomicAdd(ap + 2, coef * v.z);
    atomicAdd(ap + 3, coef * v.w);
}

// -------------------------- segmented reductions (batch sorted -> register acc)
__global__ void k_seg_sum(const float* __restrict__ h, const int* __restrict__ batch,
                          float* __restrict__ out) {
    int c = threadIdx.x;  // 128 channels
    int n0 = blockIdx.x * 128;
    int n1 = n0 + 128; if (n1 > N_NODES) n1 = N_NODES;
    float acc = 0.0f; int gc = -1;
    for (int n = n0; n < n1; ++n) {
        int g = batch[n];
        if (g != gc) {
            if (gc >= 0) atomicAdd(&out[gc * 128 + c], acc);
            acc = 0.0f; gc = g;
        }
        acc += h[(size_t)n * 128 + c];
    }
    if (gc >= 0) atomicAdd(&out[gc * 128 + c], acc);
}

__global__ void k_seg_var(const float* __restrict__ h, const int* __restrict__ batch,
                          const float* __restrict__ gsum, const float* __restrict__ cnt,
                          const float* __restrict__ gms, float* __restrict__ out) {
    int c = threadIdx.x;
    float msc = gms[c];
    int n0 = blockIdx.x * 128;
    int n1 = n0 + 128; if (n1 > N_NODES) n1 = N_NODES;
    float acc = 0.0f, mean = 0.0f; int gc = -1;
    for (int n = n0; n < n1; ++n) {
        int g = batch[n];
        if (g != gc) {
            if (gc >= 0) atomicAdd(&out[gc * 128 + c], acc);
            acc = 0.0f; gc = g;
            mean = gsum[g * 128 + c] / cnt[g];
        }
        float sub = h[(size_t)n * 128 + c] - msc * mean;
        acc += sub * sub;
    }
    if (gc >= 0) atomicAdd(&out[gc * 128 + c], acc);
}

__global__ void k_gnorm(const float* __restrict__ agg, const int* __restrict__ batch,
                        const float* __restrict__ gsum, const float* __restrict__ gvar,
                        const float* __restrict__ cnt, const float* __restrict__ gw,
                        const float* __restrict__ gb, const float* __restrict__ gms,
                        float* __restrict__ hout) {
    int i = blockIdx.x * 256 + threadIdx.x;
    if (i >= N_NODES * 128) return;
    int n = i >> 7, c = i & 127;
    int g = batch[n];
    float cg = cnt[g];
    float mean = gsum[g * 128 + c] / cg;
    float var  = gvar[g * 128 + c] / cg;
    float sub  = agg[i] - gms[c] * mean;
    float v = gw[c] * sub * rsqrtf(var + EPSV) + gb[c];
    hout[i] = fmaxf(v, 0.0f);
}

// ------------------------------------------------------------------- MLP head
__global__ void k_head1(const float* __restrict__ psum, const float* __restrict__ cnt,
                        const float* __restrict__ w0, const float* __restrict__ b0,
                        const float* __restrict__ bng, const float* __restrict__ bnb,
                        float* __restrict__ t1) {
    __shared__ float sp[128];
    int g = blockIdx.x, j = threadIdx.x;  // 256 threads
    if (j < 128) sp[j] = psum[g * 128 + j] / cnt[g];
    __syncthreads();
    float acc = b0[j];
    for (int k = 0; k < 128; ++k) acc += sp[k] * w0[k * 256 + j];
    float inv = rsqrtf(1.0f + EPSV);
    t1[g * 256 + j] = fmaxf(acc * inv * bng[j] + bnb[j], 0.0f);
}
__global__ void k_head2(const float* __restrict__ t1, const float* __restrict__ w1,
                        const float* __restrict__ b1, const float* __restrict__ bng,
                        const float* __restrict__ bnb, float* __restrict__ hout) {
    __shared__ float sp[256];
    int g = blockIdx.x, j = threadIdx.x;  // 128 threads
    sp[j] = t1[g * 256 + j];
    sp[j + 128] = t1[g * 256 + 128 + j];
    __syncthreads();
    float acc = b1[j];
    for (int k = 0; k < 256; ++k) acc += sp[k] * w1[k * 128 + j];
    float inv = rsqrtf(1.0f + EPSV);
    hout[g * 128 + j] = fmaxf(acc * inv * bng[j] + bnb[j], 0.0f);
}
__global__ void k_head3(const float* __restrict__ h2, const float* __restrict__ ow,
                        const float* __restrict__ ob, float* __restrict__ out) {
    int g = blockIdx.x * blockDim.x + threadIdx.x;
    if (g >= N_GRAPHS) return;
    float acc = ob[0];
    for (int k = 0; k < 128; ++k) acc += h2[g * 128 + k] * ow[k];
    out[g] = acc;
}

// ---------------------------------------------------------------------- host
extern "C" void kernel_launch(void* const* d_in, const int* in_sizes, int n_in,
                              void* d_out, int out_size, void* d_ws, size_t ws_size,
                              hipStream_t stream) {
    (void)in_sizes; (void)n_in; (void)out_size; (void)ws_size;
    const float* x       = (const float*)d_in[0];
    const float* ea      = (const float*)d_in[1];
    const float* em_w1   = (const float*)d_in[2];
    const float* em_b1   = (const float*)d_in[3];
    const float* em_w2   = (const float*)d_in[4];
    const float* em_b2   = (const float*)d_in[5];
    const float* conv_w[3] = {(const float*)d_in[6], (const float*)d_in[8], (const float*)d_in[10]};
    const float* conv_b[3] = {(const float*)d_in[7], (const float*)d_in[9], (const float*)d_in[11]};
    const float* gn_w[3]  = {(const float*)d_in[12], (const float*)d_in[15], (const float*)d_in[18]};
    const float* gn_b[3]  = {(const float*)d_in[13], (const float*)d_in[16], (const float*)d_in[19]};
    const float* gn_ms[3] = {(const float*)d_in[14], (const float*)d_in[17], (const float*)d_in[20]};
    const float* lin_w0 = (const float*)d_in[21];
    const float* lin_b0 = (const float*)d_in[22];
    const float* bn_g0  = (const float*)d_in[23];
    const float* bn_b0  = (const float*)d_in[24];
    const float* lin_w1 = (const float*)d_in[25];
    const float* lin_b1 = (const float*)d_in[26];
    const float* bn_g1  = (const float*)d_in[27];
    const float* bn_b1  = (const float*)d_in[28];
    const float* out_w  = (const float*)d_in[29];
    const float* out_b  = (const float*)d_in[30];
    const int* edge_index = (const int*)d_in[31];
    const int* batch      = (const int*)d_in[32];
    const int* src = edge_index;
    const int* tgt = edge_index + N_EDGES;

    float* ws   = (float*)d_ws;
    float* ew   = ws; ws += N_EDGES;
    float* deg  = ws; ws += N_NODES;
    float* dinv = ws; ws += N_NODES;
    float* cnt  = ws; ws += N_GRAPHS;
    float* buf0 = ws; ws += (size_t)N_NODES * 128;
    float* buf1 = ws; ws += (size_t)N_NODES * 128;
    float* gsum = ws; ws += N_GRAPHS * 128;
    float* gvar = ws; ws += N_GRAPHS * 128;
    float* psum = ws; ws += N_GRAPHS * 128;
    float* t1   = ws; ws += N_GRAPHS * 256;

    float* out0 = (float*)d_out;       // [256]   final scalar output
    float* outh = out0 + N_GRAPHS;     // [256,128] second tuple element

    // Edge weights + degrees + counts (done once, reused across layers)
    k_edge_mlp<<<(N_EDGES + 255) / 256, 256, 0, stream>>>(ea, em_w1, em_b1, em_w2, em_b2, ew);
    k_zero<<<512, 256, 0, stream>>>(deg, N_NODES);
    k_zero<<<1, 256, 0, stream>>>(cnt, N_GRAPHS);
    k_deg<<<(N_EDGES + 255) / 256, 256, 0, stream>>>(tgt, ew, deg);
    k_cnt<<<(N_NODES + 255) / 256, 256, 0, stream>>>(batch, cnt);
    k_cntfix<<<1, 256, 0, stream>>>(cnt);
    k_dinv<<<(N_NODES + 255) / 256, 256, 0, stream>>>(deg, dinv);

    const int NC      = N_NODES * 128;
    const int gblk    = (NC + 255) / 256;
    const int segblk  = (N_NODES + 127) / 128;
    const int gemmblk = (N_NODES + 63) / 64;

    // Ping-pong buffers: gemm(hin -> A); agg(A -> B); gnorm(B -> A); swap.
    const float* hin = x;
    float* A = buf0;
    float* B = buf1;
    for (int L = 0; L < 3; ++L) {
        if (L == 0)
            gemm_wmma<64><<<gemmblk, 256, 0, stream>>>(hin, conv_w[0], A, N_NODES);
        else
            gemm_wmma<128><<<gemmblk, 256, 0, stream>>>(hin, conv_w[L], A, N_NODES);
        k_agg_init<<<gblk, 256, 0, stream>>>(A, dinv, conv_b[L], B);
        k_scatter<<<(N_EDGES + 7) / 8, 256, 0, stream>>>(src, tgt, ew, dinv, A, B);
        k_zero<<<64, 256, 0, stream>>>(gsum, N_GRAPHS * 128);
        k_seg_sum<<<segblk, 128, 0, stream>>>(B, batch, gsum);
        k_zero<<<64, 256, 0, stream>>>(gvar, N_GRAPHS * 128);
        k_seg_var<<<segblk, 128, 0, stream>>>(B, batch, gsum, cnt, gn_ms[L], gvar);
        k_gnorm<<<gblk, 256, 0, stream>>>(B, batch, gsum, gvar, cnt, gn_w[L], gn_b[L], gn_ms[L], A);
        hin = A;
        float* tmp = A; A = B; B = tmp;
    }

    // Global mean pool + head
    k_zero<<<64, 256, 0, stream>>>(psum, N_GRAPHS * 128);
    k_seg_sum<<<segblk, 128, 0, stream>>>(hin, batch, psum);
    k_head1<<<N_GRAPHS, 256, 0, stream>>>(psum, cnt, lin_w0, lin_b0, bn_g0, bn_b0, t1);
    k_head2<<<N_GRAPHS, 128, 0, stream>>>(t1, lin_w1, lin_b1, bn_g1, bn_b1, outh);
    k_head3<<<1, 256, 0, stream>>>(outh, out_w, out_b, out0);
}
